// Wav2VecPredictionsModel_18382460027432
// MI455X (gfx1250) — matrix operand order repaired and checked by reading, not compiled, not run on hardware
//
#include <hip/hip_runtime.h>
#include <hip/hip_bf16.h>

// MI455X (gfx1250) implementation, bf16x3 split-precision GEMM.
//
// xp GEMM (103 GFLOP, L2-resident => matrix-pipe bound):
//   C[t,ci] = sum_j x[b,j,t] * W[j,ci],  ci = c*12+i  (W's native layout).
//   f32 WMMA is K=4/instr; bf16 WMMA is K=32/instr.  Split x = xh+xl,
//   W = Wh+Wl (bf16 RNE hi + residual lo) and compute xh*Wh + xh*Wl + xl*Wh
//   with f32 accumulation: ~fp32 accuracy at 8/3 the matrix throughput.
// Operands are pre-packed into WMMA fragment order (one-shot ~100MB pass) so
// every fragment load is a coalesced pair of global_load_b128.
// Stage 2 regroups predictions by s=t+i+1 into per-(b,s) 12x11x512 dot blocks
// from bank-padded LDS (bias folded into the xp rows).
//
// Workspace: yT 32MB | xp(1 batch) 50.3MB | Ah/Al 16MB ea | Bh/Bl 6MB ea.

#define BSZ    8
#define DIM    512
#define TSZ    2048
#define STEPS  12
#define NNEG   10
#define COPIES 11
#define NCI    (DIM * STEPS)          // 6144
#define NTB    (TSZ / 16)             // 128 t-tiles
#define NKB    (DIM / 32)             // 16 K-chunks of 32
#define NNB    (NCI / 16)             // 384 ci-tiles
#define FRAG   512                    // elements per packed fragment (32 lanes * 16)
#define PRED_ELEMS  2155824
#define LABEL_ELEMS 195984
#define A2S 516
#define B2S 516

typedef __attribute__((ext_vector_type(4)))  float  v4f;
typedef __attribute__((ext_vector_type(8)))  float  v8f;
typedef __attribute__((ext_vector_type(16))) __bf16 v16bf;

// ---- bf16 split helpers (RNE via bit math; no __bf16 scalar arithmetic) ----
__device__ __forceinline__ unsigned short bf16_rne(float v) {
    union { float f; unsigned int u; } c; c.f = v;
    unsigned int u = c.u;
    u += 0x7fffu + ((u >> 16) & 1u);
    return (unsigned short)(u >> 16);
}
__device__ __forceinline__ float bf16_to_f32(unsigned short h) {
    union { unsigned int u; float f; } c; c.u = ((unsigned int)h) << 16;
    return c.f;
}

// ---------------------------------------------------------------------------
// y (b,c,t) -> yT (b,t,c)
// ---------------------------------------------------------------------------
__global__ __launch_bounds__(256)
void transpose_y(const float* __restrict__ y, float* __restrict__ yT) {
    __shared__ float tile[32][33];
    const int b  = blockIdx.z;
    const int t0 = blockIdx.x * 32;
    const int c0 = blockIdx.y * 32;
    const int tx = threadIdx.x, ty = threadIdx.y;
    const float* yb  = y  + (size_t)b * DIM * TSZ;
    float*       yTb = yT + (size_t)b * TSZ * DIM;
#pragma unroll
    for (int k = 0; k < 32; k += 8)
        tile[ty + k][tx] = yb[(size_t)(c0 + ty + k) * TSZ + (t0 + tx)];
    __syncthreads();
#pragma unroll
    for (int k = 0; k < 32; k += 8)
        yTb[(size_t)(t0 + ty + k) * DIM + (c0 + tx)] = tile[tx][ty + k];
}

// ---------------------------------------------------------------------------
// Pack x into A-fragment order (hi+lo).  A 16x32 bf16 layout per ISA:
// lane = l16 + 16*half holds row M=l16, elements e=0..15 -> K = (e<8?e:e+8)+8*half.
// Afrag[b][tb][kb][lane*16+e] = bf16(x[b][kb*32+K][tb*16+l16]).
// One block per (b,tb,kb); 512 entries.
// ---------------------------------------------------------------------------
__global__ __launch_bounds__(256)
void pack_x(const float* __restrict__ x,
            unsigned short* __restrict__ Ah, unsigned short* __restrict__ Al) {
    const int blk = blockIdx.x;           // ((b*NTB + tb)*NKB + kb)
    const int kb  = blk % NKB;
    const int tb  = (blk / NKB) % NTB;
    const int b   = blk / (NKB * NTB);
    const size_t obase = (size_t)blk * FRAG;
#pragma unroll
    for (int k = 0; k < 2; ++k) {
        const int idx  = threadIdx.x + k * 256;     // 0..511
        const int lane = idx >> 4;
        const int e    = idx & 15;
        const int half = lane >> 4;
        const int l16  = lane & 15;
        const int K    = (e < 8 ? e : e + 8) + 8 * half;
        const float v  = x[(size_t)b * DIM * TSZ + (size_t)(kb * 32 + K) * TSZ
                           + (tb * 16 + l16)];
        const unsigned short h = bf16_rne(v);
        Ah[obase + idx] = h;
        Al[obase + idx] = bf16_rne(v - bf16_to_f32(h));
    }
}

// ---------------------------------------------------------------------------
// Pack W into B-fragment order (hi+lo).  B 32x16 bf16 layout:
// lane = l16 + 16*half holds col N=l16, elements e=0..15 -> K = (e<8?e:e+8)+8*half.
// Bfrag[kb][nb][lane*16+e] = bf16(W[kb*32+K][nb*16+l16]).
// ---------------------------------------------------------------------------
__global__ __launch_bounds__(256)
void pack_w(const float* __restrict__ W,
            unsigned short* __restrict__ Bh, unsigned short* __restrict__ Bl) {
    const int blk = blockIdx.x;           // kb*NNB + nb
    const int nb  = blk % NNB;
    const int kb  = blk / NNB;
    const size_t obase = (size_t)blk * FRAG;
#pragma unroll
    for (int k = 0; k < 2; ++k) {
        const int idx  = threadIdx.x + k * 256;
        const int lane = idx >> 4;
        const int e    = idx & 15;
        const int half = lane >> 4;
        const int l16  = lane & 15;
        const int K    = (e < 8 ? e : e + 8) + 8 * half;
        const float v  = W[(size_t)(kb * 32 + K) * NCI + (nb * 16 + l16)];
        const unsigned short h = bf16_rne(v);
        Bh[obase + idx] = h;
        Bl[obase + idx] = bf16_rne(v - bf16_to_f32(h));
    }
}

// ---------------------------------------------------------------------------
// bf16x3 WMMA GEMM for one batch.  Wave tile 32M x 64N (8 v8f accumulators),
// workgroup 8 waves as 2(M) x 4(N) => 64M x 256N.  K loop: 16 chunks of 32.
// Per chunk per wave: 12 coalesced 32B fragment loads, 24 bf16 WMMAs.
// ---------------------------------------------------------------------------
__global__ __launch_bounds__(256)
void gemm_bf16x3(const unsigned short* __restrict__ Ahb,  // [NTB][NKB][FRAG] (this b)
                 const unsigned short* __restrict__ Alb,
                 const unsigned short* __restrict__ Bh,   // [NKB][NNB][FRAG]
                 const unsigned short* __restrict__ Bl,
                 float* __restrict__ xpws)                // (TSZ, NCI)
{
    const int lane = threadIdx.x & 31;
    const int wave = threadIdx.x >> 5;
    const int mw   = wave >> 2;          // 0..1
    const int nw   = wave & 3;           // 0..3
    const int half = lane >> 4;
    const int l16  = lane & 15;

    const int tb0 = blockIdx.y * 4  + mw * 2;   // first of 2 m-tiles
    const int nb0 = blockIdx.x * 16 + nw * 4;   // first of 4 n-tiles

    v8f acc[2][4];
    const v8f vzero = {0.f, 0.f, 0.f, 0.f, 0.f, 0.f, 0.f, 0.f};
#pragma unroll
    for (int mt = 0; mt < 2; ++mt)
#pragma unroll
        for (int nt = 0; nt < 4; ++nt) acc[mt][nt] = vzero;

    for (int kb = 0; kb < NKB; ++kb) {
        v16bf ah[2], al[2];
#pragma unroll
        for (int mt = 0; mt < 2; ++mt) {
            const size_t base = ((size_t)(tb0 + mt) * NKB + kb) * FRAG + lane * 16;
            ah[mt] = *reinterpret_cast<const v16bf*>(Ahb + base);
            al[mt] = *reinterpret_cast<const v16bf*>(Alb + base);
        }
        v16bf bh[4], bl[4];
#pragma unroll
        for (int nt = 0; nt < 4; ++nt) {
            const size_t base = ((size_t)kb * NNB + (nb0 + nt)) * FRAG + lane * 16;
            bh[nt] = *reinterpret_cast<const v16bf*>(Bh + base);
            bl[nt] = *reinterpret_cast<const v16bf*>(Bl + base);
        }
#pragma unroll
        for (int mt = 0; mt < 2; ++mt) {
#pragma unroll
            for (int nt = 0; nt < 4; ++nt) {
                acc[mt][nt] = __builtin_amdgcn_wmma_f32_16x16x32_bf16(
                    false, ah[mt], false, bh[nt], (short)0, acc[mt][nt], false, false);
                acc[mt][nt] = __builtin_amdgcn_wmma_f32_16x16x32_bf16(
                    false, ah[mt], false, bl[nt], (short)0, acc[mt][nt], false, false);
                acc[mt][nt] = __builtin_amdgcn_wmma_f32_16x16x32_bf16(
                    false, al[mt], false, bh[nt], (short)0, acc[mt][nt], false, false);
            }
        }
    }

    // C layout: VGPR r -> row r + 8*half, col l16.  Coalesced b32 stores.
#pragma unroll
    for (int mt = 0; mt < 2; ++mt) {
#pragma unroll
        for (int nt = 0; nt < 4; ++nt) {
            const int ci = (nb0 + nt) * 16 + l16;
#pragma unroll
            for (int r = 0; r < 8; ++r) {
                const int t = (tb0 + mt) * 16 + r + 8 * half;
                xpws[(size_t)t * NCI + ci] = acc[mt][nt][r];
            }
        }
    }
}

// ---------------------------------------------------------------------------
// Stage 2: one block per (b, s), s = t+i+1 in [1, 2047].
// ---------------------------------------------------------------------------
__global__ __launch_bounds__(256)
void stage2(const float* __restrict__ xpws, const float* __restrict__ yT,
            const int* __restrict__ neg, const float* __restrict__ bias,
            float* __restrict__ out, int b)
{
    __shared__ __align__(16) float A2[STEPS * A2S];
    __shared__ __align__(16) float B2[COPIES * B2S];
    const int s   = blockIdx.x + 1;
    const int tid = threadIdx.x;

    for (int idx = tid; idx < STEPS * DIM; idx += 256) {
        const int i = idx >> 9;
        const int c = idx & 511;
        const int t = s - 1 - i;
        float v = (t >= 0) ? xpws[(size_t)t * NCI + c * STEPS + i] : 0.f;
        A2[i * A2S + c] = v + bias[c];
    }
    for (int idx = tid; idx < COPIES * DIM; idx += 256) {
        const int n = idx >> 9;
        const int c = idx & 511;
        int src;
        if (n == 0) src = b * TSZ + s;
        else        src = neg[(size_t)b * (NNEG * TSZ) + (n - 1) * TSZ + s];
        B2[n * B2S + c] = yT[(size_t)src * DIM + c];
    }
    __syncthreads();

    if (tid < STEPS * COPIES) {
        const int i = tid / COPIES;
        const int n = tid % COPIES;
        const int t = s - 1 - i;
        if (t >= 0) {
            const v4f* pa = (const v4f*)(A2 + i * A2S);
            const v4f* pb = (const v4f*)(B2 + n * B2S);
            float accv = 0.f;
#pragma unroll 4
            for (int c4 = 0; c4 < DIM / 4; ++c4) {
                const v4f av = pa[c4];
                const v4f bv = pb[c4];
                accv += av.x * bv.x + av.y * bv.y + av.z * bv.z + av.w * bv.w;
            }
            const int base_i = 88 * (2047 * i - (i * (i - 1)) / 2);
            out[base_i + (t * 8 + b) * COPIES + n] = accv;
        }
    }
}

__global__ void zero_labels(float* __restrict__ out) {
    const int i = blockIdx.x * blockDim.x + threadIdx.x;
    if (i < LABEL_ELEMS) out[PRED_ELEMS + i] = 0.f;
}

extern "C" void kernel_launch(void* const* d_in, const int* in_sizes, int n_in,
                              void* d_out, int out_size, void* d_ws, size_t ws_size,
                              hipStream_t stream) {
    const float* x    = (const float*)d_in[0];
    const float* y    = (const float*)d_in[1];
    const float* W    = (const float*)d_in[2];
    const float* bias = (const float*)d_in[3];
    const int*   neg  = (const int*)  d_in[4];
    float* out = (float*)d_out;

    char* ws = (char*)d_ws;
    float* yT = (float*)ws;                                  ws += (size_t)BSZ * TSZ * DIM * 4;   // 32MB
    float* xpws = (float*)ws;                                ws += (size_t)TSZ * NCI * 4;         // 50.3MB
    unsigned short* Ah = (unsigned short*)ws;                ws += (size_t)BSZ * NTB * NKB * FRAG * 2; // 16MB
    unsigned short* Al = (unsigned short*)ws;                ws += (size_t)BSZ * NTB * NKB * FRAG * 2; // 16MB
    unsigned short* Bh = (unsigned short*)ws;                ws += (size_t)NKB * NNB * FRAG * 2;  // 6MB
    unsigned short* Bl = (unsigned short*)ws;                                                     // 6MB

    transpose_y<<<dim3(TSZ / 32, DIM / 32, BSZ), dim3(32, 8), 0, stream>>>(y, yT);
    zero_labels<<<(LABEL_ELEMS + 255) / 256, 256, 0, stream>>>(out);
    pack_w<<<NKB * NNB, 256, 0, stream>>>(W, Bh, Bl);
    pack_x<<<BSZ * NTB * NKB, 256, 0, stream>>>(x, Ah, Al);

    for (int b = 0; b < BSZ; ++b) {
        const size_t abase = (size_t)b * NTB * NKB * FRAG;
        gemm_bf16x3<<<dim3(NNB / 16, NTB / 4), 256, 0, stream>>>(
            Ah + abase, Al + abase, Bh, Bl, xpws);
        stage2<<<TSZ - 1, 256, 0, stream>>>(xpws, yT, neg, bias, out, b);
    }
}
